// RelPosEmbConformerBlock_66271345377654
// MI455X (gfx1250) — matrix operand order repaired and checked by reading, not compiled, not run on hardware
//
#include <hip/hip_runtime.h>
#include <hip/hip_bf16.h>

// ---------------------------------------------------------------------------
// RelPosEmbConformerBlock for MI455X (gfx1250, wave32, WMMA bf16->f32,
// async global->LDS staging). B=8, T=1024, D=512, H=8, DK=64, conv K=33
// ---------------------------------------------------------------------------

typedef __bf16 bf16_t;
typedef __attribute__((ext_vector_type(16))) __bf16 v16bf;
typedef __attribute__((ext_vector_type(8)))  __bf16 v8bf;
typedef __attribute__((ext_vector_type(8)))  float  v8f;

#define CB 8
#define CT 1024
#define CD 512
#define CH 8
#define CDK 64

enum { EPI_NONE = 0, EPI_SILU = 1, EPI_ADD = 2, EPI_HALF = 3 };

#if defined(__HIP_DEVICE_COMPILE__) && __has_builtin(__builtin_amdgcn_global_load_async_to_lds_b128)
#define ASYNC_LDS 1
typedef __attribute__((ext_vector_type(4))) int v4i_t;
typedef __attribute__((address_space(1))) v4i_t* gv4i_t;
typedef __attribute__((address_space(3))) v4i_t* lv4i_t;
#else
#define ASYNC_LDS 0
#endif

// copy 8 bf16 (16B) global -> LDS, async (ASYNCcnt) if available
__device__ __forceinline__ void stage16(bf16_t* dst, const bf16_t* src) {
#if ASYNC_LDS
  __builtin_amdgcn_global_load_async_to_lds_b128(
      (gv4i_t)(void*)const_cast<bf16_t*>(src), (lv4i_t)(void*)dst, 0, 0);
#else
  *(v8bf*)dst = *(const v8bf*)src;
#endif
}

__device__ __forceinline__ void async_wait_all() {
#if ASYNC_LDS
#if __has_builtin(__builtin_amdgcn_s_wait_asynccnt)
  __builtin_amdgcn_s_wait_asynccnt(0);
#else
  asm volatile("s_wait_asynccnt 0" ::: "memory");
#endif
#endif
}

__device__ __forceinline__ v8f v8f_zero() {
  v8f z;
#pragma unroll
  for (int i = 0; i < 8; ++i) z[i] = 0.f;
  return z;
}

__device__ __forceinline__ v8f wmma_bf16f32(v16bf a, v16bf b, v8f c) {
  // D = A(16x32 bf16) x B(32x16 bf16) + C(16x16 f32)
  return __builtin_amdgcn_wmma_f32_16x16x32_bf16(false, a, false, b, (short)0, c,
                                                 false, false);
}

// A-operand fragment (16 rows x 32 k) from row-major [row][k] storage.
// ISA layout: lane<16 -> k {klo..klo+7, 16+klo..+7}, klo=(lane>>4)*8.
__device__ __forceinline__ v16bf fragA(const bf16_t* p_, int row, int koff, int stride) {
  int lane = threadIdx.x & 31;
  const bf16_t* p = p_ + (size_t)(row + (lane & 15)) * stride + koff + ((lane >> 4) << 3);
  v8bf lo = *(const v8bf*)p;
  v8bf hi = *(const v8bf*)(p + 16);
  v16bf f;
#pragma unroll
  for (int i = 0; i < 8; ++i) { f[i] = lo[i]; f[i + 8] = hi[i]; }
  return f;
}

// B-operand fragment (32 k x 16 n) from LDS stored [n][k] (n-major).
// ISA layout: lane<16 -> k {klo..klo+7, 16+klo..+7}, klo=(lane>>4)*8, n = lane&15.
__device__ __forceinline__ v16bf fragB(const bf16_t* p_, int ncol, int koff, int stride) {
  int lane = threadIdx.x & 31;
  const bf16_t* p = p_ + (size_t)(ncol + (lane & 15)) * stride + koff + ((lane >> 4) << 3);
  v8bf lo = *(const v8bf*)p;
  v8bf hi = *(const v8bf*)(p + 16);
  v16bf f;
#pragma unroll
  for (int i = 0; i < 8; ++i) { f[i] = lo[i]; f[i + 8] = hi[i]; }
  return f;
}

// ---------------------------------------------------------------------------
// GEMM: C[M,N] = epilogue(A_bf16[M,K] @ Wt_bf16[N,K]^T + bias, res)
// Weights pre-packed [N,K] so both tile fills are contiguous b128 async copies.
// block tile 128x128, BK=32, double-buffered LDS, 256 threads (8 waves, 2x4).
// ---------------------------------------------------------------------------
template <int EPI>
__global__ void __launch_bounds__(256)
gemm_bf16_kernel(const bf16_t* __restrict__ A, const bf16_t* __restrict__ Bt,
                 const float* __restrict__ bias, const float* __restrict__ res,
                 float* __restrict__ C, int M, int N, int K) {
  __shared__ alignas(16) bf16_t As[2][128 * 40];
  __shared__ alignas(16) bf16_t Bs[2][128 * 40];

  const int tid = threadIdx.x;
  const int lane = tid & 31;
  const int wid = tid >> 5;
  const int wm = wid >> 2;  // 0..1
  const int wn = wid & 3;   // 0..3
  const int m0 = blockIdx.y * 128;
  const int n0 = blockIdx.x * 128;

  v8f acc[4][2];
#pragma unroll
  for (int i = 0; i < 4; ++i)
#pragma unroll
    for (int j = 0; j < 2; ++j) acc[i][j] = v8f_zero();

  const int row = tid >> 1, kh = (tid & 1) * 16;
  const bf16_t* aSrc = A + (size_t)(m0 + row) * K + kh;
  const bf16_t* bSrc = Bt + (size_t)(n0 + row) * K + kh;

  auto stageTile = [&](int kt, int buf) {
    const bf16_t* sa = aSrc + kt * 32;
    bf16_t* da = &As[buf][row * 40 + kh];
    stage16(da, sa);
    stage16(da + 8, sa + 8);
    const bf16_t* sb = bSrc + kt * 32;
    bf16_t* db = &Bs[buf][row * 40 + kh];
    stage16(db, sb);
    stage16(db + 8, sb + 8);
  };

  const int NK = K / 32;
  stageTile(0, 0);
  for (int kt = 0; kt < NK; ++kt) {
    async_wait_all();
    __syncthreads();
    const int cur = kt & 1;
    if (kt + 1 < NK) stageTile(kt + 1, cur ^ 1);
    if (kt + 2 < NK) __builtin_prefetch(aSrc + (kt + 2) * 32, 0, 0);

    v16bf af[4], bfr[2];
#pragma unroll
    for (int i = 0; i < 4; ++i) af[i] = fragA(As[cur], wm * 64 + i * 16, 0, 40);
#pragma unroll
    for (int j = 0; j < 2; ++j) bfr[j] = fragB(Bs[cur], wn * 32 + j * 16, 0, 40);
#pragma unroll
    for (int i = 0; i < 4; ++i)
#pragma unroll
      for (int j = 0; j < 2; ++j) acc[i][j] = wmma_bf16f32(af[i], bfr[j], acc[i][j]);
  }

#pragma unroll
  for (int i = 0; i < 4; ++i) {
#pragma unroll
    for (int j = 0; j < 2; ++j) {
      const int col = n0 + wn * 32 + j * 16 + (lane & 15);
      const int rbase = m0 + wm * 64 + i * 16 + ((lane >> 4) << 3);
      const float bv = bias ? bias[col] : 0.f;
#pragma unroll
      for (int r = 0; r < 8; ++r) {
        const size_t o = (size_t)(rbase + r) * N + col;
        float v = acc[i][j][r] + bv;
        if (EPI == EPI_SILU) v = v / (1.f + __expf(-v));
        else if (EPI == EPI_ADD) v += res[o];
        else if (EPI == EPI_HALF) v = res[o] + 0.5f * v;
        C[o] = v;
      }
    }
  }
}

// ---------------------------------------------------------------------------
// LayerNorm over last dim L, one block (256 thr) per row, optional SiLU.
// ---------------------------------------------------------------------------
template <int L, bool SILU>
__global__ void __launch_bounds__(256)
ln_kernel(const float* __restrict__ x, const float* __restrict__ g,
          const float* __restrict__ b, bf16_t* __restrict__ obf,
          float* __restrict__ of) {
  constexpr int PER = L / 256;
  const int row = blockIdx.x;
  const int tid = threadIdx.x;
  const float* xr = x + (size_t)row * L;
  float v[PER];
  float s = 0.f, s2 = 0.f;
#pragma unroll
  for (int i = 0; i < PER; ++i) {
    v[i] = xr[tid + i * 256];
    s += v[i];
    s2 += v[i] * v[i];
  }
#pragma unroll
  for (int off = 16; off >= 1; off >>= 1) {
    s += __shfl_xor(s, off, 32);
    s2 += __shfl_xor(s2, off, 32);
  }
  __shared__ float red[16];
  if ((tid & 31) == 0) { red[tid >> 5] = s; red[8 + (tid >> 5)] = s2; }
  __syncthreads();
  s = 0.f; s2 = 0.f;
#pragma unroll
  for (int i = 0; i < 8; ++i) { s += red[i]; s2 += red[8 + i]; }
  const float mean = s / (float)L;
  const float rstd = rsqrtf(s2 / (float)L - mean * mean + 1e-5f);
#pragma unroll
  for (int i = 0; i < PER; ++i) {
    const int c = tid + i * 256;
    float y = (v[i] - mean) * rstd * g[c] + b[c];
    if (SILU) y = y / (1.f + __expf(-y));
    if (obf) obf[(size_t)row * L + c] = (bf16_t)y;
    if (of) of[(size_t)row * L + c] = y;
  }
}

__global__ void cvt_bf16_kernel(const float* __restrict__ in,
                                bf16_t* __restrict__ out, size_t n) {
  size_t i = (size_t)blockIdx.x * blockDim.x + threadIdx.x;
  const size_t stride = (size_t)gridDim.x * blockDim.x;
  for (; i < n; i += stride) out[i] = (bf16_t)in[i];
}

// weight pack: f32 [K,N] -> bf16 [N,K] (transposed once per launch)
__global__ void packw_kernel(const float* __restrict__ in,
                             bf16_t* __restrict__ out, int K, int N, size_t n) {
  const size_t i = (size_t)blockIdx.x * blockDim.x + threadIdx.x;
  if (i >= n) return;
  const int k = (int)(i / N), nn = (int)(i % N);
  out[(size_t)nn * K + k] = (bf16_t)in[i];
}

// GLU: [M,2048] -> a*sigmoid(g) [M,1024]
__global__ void glu_kernel(const float* __restrict__ h, float* __restrict__ out,
                           size_t n) {
  const size_t i = (size_t)blockIdx.x * blockDim.x + threadIdx.x;
  if (i >= n) return;
  const size_t r = i >> 10;
  const int c = (int)(i & 1023);
  const float a = h[r * 2048 + c];
  const float gg = h[r * 2048 + 1024 + c];
  out[i] = a / (1.f + __expf(-gg));
}

// depthwise conv over T, C=1024, K=33, pad 16, weight [C][33]
__global__ void dwconv_kernel(const float* __restrict__ in,
                              const float* __restrict__ w,
                              float* __restrict__ out) {
  const size_t i = (size_t)blockIdx.x * blockDim.x + threadIdx.x;
  const int c = (int)(i & 1023);
  const size_t bt = i >> 10;
  const int t = (int)(bt & (CT - 1));
  const size_t bb = bt >> 10;
  float acc = 0.f;
#pragma unroll
  for (int k = 0; k < 33; ++k) {
    const int tt = t + k - 16;
    if (tt >= 0 && tt < CT) acc += in[((bb << 10) + (size_t)tt) * 1024 + c] * w[c * 33 + k];
  }
  out[i] = acc;
}

// [rows, D] f32 -> [B,H,T,DK] bf16, optional per-(h,dk) bias, optional 2nd copy
__global__ void heads_kernel(const float* __restrict__ src,
                             const float* __restrict__ bias1,
                             bf16_t* __restrict__ dst1,
                             const float* __restrict__ bias2,
                             bf16_t* __restrict__ dst2, size_t n) {
  const size_t i = (size_t)blockIdx.x * blockDim.x + threadIdx.x;
  if (i >= n) return;
  const int col = (int)(i & (CD - 1));
  const size_t row = i >> 9;
  const int t = (int)(row & (CT - 1));
  const size_t bb = row >> 10;
  const int h = col >> 6, dk = col & 63;
  const size_t d = ((bb * CH + h) * CT + t) * CDK + dk;
  const float v = src[i];
  dst1[d] = (bf16_t)(v + (bias1 ? bias1[col] : 0.f));
  if (dst2) dst2[d] = (bf16_t)(v + (bias2 ? bias2[col] : 0.f));
}

// [B*T, D] f32 -> [B,H,DK,T] bf16 (V pre-transposed for attention B-operand)
__global__ void headsT_kernel(const float* __restrict__ src,
                              bf16_t* __restrict__ dst, size_t n) {
  const size_t i = (size_t)blockIdx.x * blockDim.x + threadIdx.x;
  if (i >= n) return;
  const int col = (int)(i & (CD - 1));
  const size_t row = i >> 9;
  const int t = (int)(row & (CT - 1));
  const size_t bb = row >> 10;
  const int h = col >> 6, dk = col & 63;
  dst[((bb * CH + h) * CDK + dk) * CT + t] = (bf16_t)src[i];
}

// ---------------------------------------------------------------------------
// Fused rel-pos flash attention.
// grid (T/64, B*H), 128 threads = 4 waves; wave w owns 16 q rows.
// scores = scale*(qu.K^T) + scale*rel_shift(qv.p); online softmax; O = P.V
// ---------------------------------------------------------------------------
__global__ void __launch_bounds__(128)
attn_kernel(const bf16_t* __restrict__ QU, const bf16_t* __restrict__ QV,
            const bf16_t* __restrict__ Kb, const bf16_t* __restrict__ Vt,
            const bf16_t* __restrict__ Pb, bf16_t* __restrict__ Ob) {
  const int T = CT;
  const int bh = blockIdx.y;
  const int h = bh & (CH - 1);
  const int bb = bh >> 3;
  const int q0 = blockIdx.x * 64;
  const int tid = threadIdx.x, lane = tid & 31, w = tid >> 5;
  const int nl = lane & 15, hi = lane >> 4;

  const bf16_t* QUh = QU + (size_t)bh * T * CDK;
  const bf16_t* QVh = QV + (size_t)bh * T * CDK;
  const bf16_t* Kh = Kb + (size_t)bh * T * CDK;   // [key][dk]
  const bf16_t* Vth = Vt + (size_t)bh * CDK * T;  // [dk][t]  (pre-transposed)
  const bf16_t* Ph = Pb + (size_t)h * T * CDK;

  __shared__ alignas(16) bf16_t Ks[64 * 72];      // [key][dk]
  __shared__ alignas(16) bf16_t Vs[64 * 72];      // [dk][key]
  __shared__ alignas(16) bf16_t Ps[128 * 72];     // [c][dk], c = j - Jbase
  __shared__ alignas(16) bf16_t BDs[4][16 * 80];  // per-wave bd staging
  __shared__ alignas(16) bf16_t SPs[4][16 * 72];  // per-wave P staging

  const int qrow = q0 + w * 16;
  v16bf qu[2], qv[2];
#pragma unroll
  for (int c = 0; c < 2; ++c) {
    qu[c] = fragA(QUh, qrow, c * 32, CDK);
    qv[c] = fragA(QVh, qrow, c * 32, CDK);
  }

  v8f Oacc[4];
#pragma unroll
  for (int i = 0; i < 4; ++i) Oacc[i] = v8f_zero();
  float mst[8], lst[8];
#pragma unroll
  for (int r = 0; r < 8; ++r) { mst[r] = -1e30f; lst[r] = 0.f; }
  const float scale = 0.125f;  // 1/sqrt(64)

  for (int k0 = 0; k0 < T; k0 += 64) {
    __syncthreads();
    {  // cooperative async tile staging (all contiguous 16B copies)
      const int row = tid >> 1, seg = (tid & 1) * 32;
      const bf16_t* ksrc = Kh + (size_t)(k0 + row) * CDK + seg;
      bf16_t* kdst = Ks + row * 72 + seg;
#pragma unroll
      for (int i = 0; i < 4; ++i) stage16(kdst + i * 8, ksrc + i * 8);
      const bf16_t* vsrc = Vth + (size_t)row * T + k0 + seg;  // row = dk here
      bf16_t* vdst = Vs + row * 72 + seg;
#pragma unroll
      for (int i = 0; i < 4; ++i) stage16(vdst + i * 8, vsrc + i * 8);
      const int j = (T - 1) + k0 - q0 - 64 + tid;  // c = tid
      if (j >= 0 && j < T) {
        const bf16_t* ps = Ph + (size_t)j * CDK;
#pragma unroll
        for (int i = 0; i < 8; ++i) stage16(Ps + tid * 72 + i * 8, ps + i * 8);
      } else {
        v8bf z;
#pragma unroll
        for (int q = 0; q < 8; ++q) z[q] = (bf16_t)0.f;
#pragma unroll
        for (int i = 0; i < 8; ++i) *(v8bf*)(Ps + tid * 72 + i * 8) = z;
      }
    }
    async_wait_all();
    __syncthreads();

    // S = qu @ K^T  (4 n-frags x 2 dk chunks)
    v8f S[4];
#pragma unroll
    for (int ni = 0; ni < 4; ++ni) {
      v8f s = v8f_zero();
#pragma unroll
      for (int c = 0; c < 2; ++c) s = wmma_bf16f32(qu[c], fragB(Ks, ni * 16, c * 32, 72), s);
      S[ni] = s;
    }
    // bd = qv @ p window; wave w needs Ps cols [(3-w)*16 .. (8-w)*16)
#pragma unroll
    for (int fc = 0; fc < 5; ++fc) {
      const int ncb = (3 - w + fc) * 16;
      v8f bd = v8f_zero();
#pragma unroll
      for (int c = 0; c < 2; ++c) bd = wmma_bf16f32(qv[c], fragB(Ps, ncb, c * 32, 72), bd);
#pragma unroll
      for (int r = 0; r < 8; ++r)
        BDs[w][(r + hi * 8) * 80 + fc * 16 + nl] = (bf16_t)bd[r];
    }
    asm volatile("s_wait_dscnt 0" ::: "memory");

    // combine (rel-shift index: stored col = 16 + n - m, wave-independent)
    float rmax[8];
#pragma unroll
    for (int r = 0; r < 8; ++r) rmax[r] = -1e30f;
#pragma unroll
    for (int ni = 0; ni < 4; ++ni) {
#pragma unroll
      for (int r = 0; r < 8; ++r) {
        const int m = r + hi * 8;
        const int n = ni * 16 + nl;
        const float sv = scale * (S[ni][r] + (float)BDs[w][m * 80 + 16 + n - m]);
        S[ni][r] = sv;
        rmax[r] = fmaxf(rmax[r], sv);
      }
    }
#pragma unroll
    for (int r = 0; r < 8; ++r)
#pragma unroll
      for (int off = 8; off >= 1; off >>= 1)
        rmax[r] = fmaxf(rmax[r], __shfl_xor(rmax[r], off, 32));

    float alpha[8], rsum[8];
#pragma unroll
    for (int r = 0; r < 8; ++r) {
      const float mn = fmaxf(mst[r], rmax[r]);
      alpha[r] = __expf(mst[r] - mn);
      mst[r] = mn;
      rsum[r] = 0.f;
    }
#pragma unroll
    for (int ni = 0; ni < 4; ++ni)
#pragma unroll
      for (int r = 0; r < 8; ++r) {
        const float p = __expf(S[ni][r] - mst[r]);
        S[ni][r] = p;
        rsum[r] += p;
      }
#pragma unroll
    for (int r = 0; r < 8; ++r) {
#pragma unroll
      for (int off = 8; off >= 1; off >>= 1) rsum[r] += __shfl_xor(rsum[r], off, 32);
      lst[r] = lst[r] * alpha[r] + rsum[r];
    }
    // stage P (C-layout -> LDS row-major) for A-fragment reload
#pragma unroll
    for (int ni = 0; ni < 4; ++ni)
#pragma unroll
      for (int r = 0; r < 8; ++r)
        SPs[w][(r + hi * 8) * 72 + ni * 16 + nl] = (bf16_t)S[ni][r];
    asm volatile("s_wait_dscnt 0" ::: "memory");

    v16bf pa[2];
#pragma unroll
    for (int c = 0; c < 2; ++c) pa[c] = fragA(SPs[w], 0, c * 32, 72);
#pragma unroll
    for (int nf = 0; nf < 4; ++nf) {
#pragma unroll
      for (int r = 0; r < 8; ++r) Oacc[nf][r] *= alpha[r];
#pragma unroll
      for (int c = 0; c < 2; ++c)
        Oacc[nf] = wmma_bf16f32(pa[c], fragB(Vs, nf * 16, c * 32, 72), Oacc[nf]);
    }
  }

  // write O back to [B,T,D] bf16
#pragma unroll
  for (int nf = 0; nf < 4; ++nf) {
#pragma unroll
    for (int r = 0; r < 8; ++r) {
      const int m = r + hi * 8;
      const int dk = nf * 16 + nl;
      Ob[((size_t)bb * CT + (qrow + m)) * CD + h * CDK + dk] =
          (bf16_t)(Oacc[nf][r] / lst[r]);
    }
  }
}

// ---------------------------------------------------------------------------
// Host orchestration
// ---------------------------------------------------------------------------
extern "C" void kernel_launch(void* const* d_in, const int* in_sizes, int n_in,
                              void* d_out, int out_size, void* d_ws,
                              size_t ws_size, hipStream_t stream) {
  (void)in_sizes; (void)n_in; (void)out_size; (void)ws_size;
  const int M = CB * CT;  // 8192

  const float* x = (const float*)d_in[0];
  const float* pos_emb = (const float*)d_in[1];
  const float* ln1_g = (const float*)d_in[2];
  const float* ln1_b = (const float*)d_in[3];
  const float* ffn1_b1 = (const float*)d_in[5];
  const float* ffn1_b2 = (const float*)d_in[7];
  const float* lnq_g = (const float*)d_in[8];
  const float* lnq_b = (const float*)d_in[9];
  const float* lnk_g = (const float*)d_in[10];
  const float* lnk_b = (const float*)d_in[11];
  const float* lnv_g = (const float*)d_in[12];
  const float* lnv_b = (const float*)d_in[13];
  const float* pbu = (const float*)d_in[18];
  const float* pbv = (const float*)d_in[19];
  const float* cln_g = (const float*)d_in[21];
  const float* cln_b = (const float*)d_in[22];
  const float* dw_w = (const float*)d_in[24];
  const float* bn_g = (const float*)d_in[25];
  const float* bn_b = (const float*)d_in[26];
  const float* ln2_g = (const float*)d_in[28];
  const float* ln2_b = (const float*)d_in[29];
  const float* ffn2_b1 = (const float*)d_in[31];
  const float* ffn2_b2 = (const float*)d_in[33];
  const float* lnf_g = (const float*)d_in[34];
  const float* lnf_b = (const float*)d_in[35];

  char* base = (char*)d_ws;
  size_t off = 0;
  auto alloc = [&](size_t bytes) -> void* {
    void* p = base + off;
    off += (bytes + 255) & ~(size_t)255;
    return p;
  };

  bf16_t* w1b = (bf16_t*)alloc((size_t)512 * 2048 * 2);
  bf16_t* w2b = (bf16_t*)alloc((size_t)2048 * 512 * 2);
  bf16_t* wqb = (bf16_t*)alloc((size_t)512 * 512 * 2);
  bf16_t* wkb = (bf16_t*)alloc((size_t)512 * 512 * 2);
  bf16_t* wvb = (bf16_t*)alloc((size_t)512 * 512 * 2);
  bf16_t* wpb = (bf16_t*)alloc((size_t)512 * 512 * 2);
  bf16_t* wfb = (bf16_t*)alloc((size_t)512 * 512 * 2);
  bf16_t* pw1b = (bf16_t*)alloc((size_t)512 * 2048 * 2);
  bf16_t* pw2b = (bf16_t*)alloc((size_t)1024 * 512 * 2);
  bf16_t* f21b = (bf16_t*)alloc((size_t)512 * 2048 * 2);
  bf16_t* f22b = (bf16_t*)alloc((size_t)2048 * 512 * 2);
  bf16_t* abf = (bf16_t*)alloc((size_t)M * 512 * 2);
  bf16_t* pebf = (bf16_t*)alloc((size_t)1024 * 512 * 2);
  float* hf = (float*)alloc((size_t)M * 2048 * 4);
  bf16_t* hb = (bf16_t*)alloc((size_t)M * 2048 * 2);
  float* outA = (float*)alloc((size_t)M * 512 * 4);
  float* outB = (float*)alloc((size_t)M * 512 * 4);
  float* qf = (float*)alloc((size_t)M * 512 * 4);
  float* kf = (float*)alloc((size_t)M * 512 * 4);
  float* vf = (float*)alloc((size_t)M * 512 * 4);
  float* pf = (float*)alloc((size_t)1024 * 512 * 4);
  bf16_t* qub = (bf16_t*)alloc((size_t)M * 512 * 2);
  bf16_t* qvb = (bf16_t*)alloc((size_t)M * 512 * 2);
  bf16_t* kbb = (bf16_t*)alloc((size_t)M * 512 * 2);
  bf16_t* vbb = (bf16_t*)alloc((size_t)M * 512 * 2);  // [B,H,DK,T]
  bf16_t* pbb = (bf16_t*)alloc((size_t)1024 * 512 * 2);
  bf16_t* aob = (bf16_t*)alloc((size_t)M * 512 * 2);
  float* dcf = (float*)alloc((size_t)M * 1024 * 4);
  bf16_t* cbb = (bf16_t*)alloc((size_t)M * 1024 * 2);
  float* gluf = (float*)hb;  // reuse (sequential stages)

  auto cvt = [&](const float* s, bf16_t* dd, size_t n) {
    cvt_bf16_kernel<<<dim3((unsigned)((n + 255) / 256)), dim3(256), 0, stream>>>(s, dd, n);
  };
  auto packw = [&](const float* s, bf16_t* dd, int K, int N) {
    const size_t n = (size_t)K * N;
    packw_kernel<<<dim3((unsigned)((n + 255) / 256)), dim3(256), 0, stream>>>(s, dd, K, N, n);
  };
  auto gemm = [&](int epi, const bf16_t* A, const bf16_t* Bw, const float* bias,
                  const float* res, float* C, int Mm, int Nn, int Kk) {
    dim3 g(Nn / 128, Mm / 128), blk(256);
    if (epi == EPI_NONE)
      gemm_bf16_kernel<EPI_NONE><<<g, blk, 0, stream>>>(A, Bw, bias, res, C, Mm, Nn, Kk);
    else if (epi == EPI_SILU)
      gemm_bf16_kernel<EPI_SILU><<<g, blk, 0, stream>>>(A, Bw, bias, res, C, Mm, Nn, Kk);
    else if (epi == EPI_ADD)
      gemm_bf16_kernel<EPI_ADD><<<g, blk, 0, stream>>>(A, Bw, bias, res, C, Mm, Nn, Kk);
    else
      gemm_bf16_kernel<EPI_HALF><<<g, blk, 0, stream>>>(A, Bw, bias, res, C, Mm, Nn, Kk);
  };

  // weights -> bf16 [N,K] (transposed pack)
  packw((const float*)d_in[4], w1b, 512, 2048);
  packw((const float*)d_in[6], w2b, 2048, 512);
  packw((const float*)d_in[14], wqb, 512, 512);
  packw((const float*)d_in[15], wkb, 512, 512);
  packw((const float*)d_in[16], wvb, 512, 512);
  packw((const float*)d_in[17], wpb, 512, 512);
  packw((const float*)d_in[20], wfb, 512, 512);
  packw((const float*)d_in[23], pw1b, 512, 2048);
  packw((const float*)d_in[27], pw2b, 1024, 512);
  packw((const float*)d_in[30], f21b, 512, 2048);
  packw((const float*)d_in[32], f22b, 2048, 512);
  cvt(pos_emb, pebf, (size_t)1024 * 512);

  // ---- FFN1 (half-step): out0 = x + 0.5*ffn1(x) ----
  ln_kernel<512, false><<<M, 256, 0, stream>>>(x, ln1_g, ln1_b, abf, nullptr);
  gemm(EPI_SILU, abf, w1b, ffn1_b1, nullptr, hf, M, 2048, 512);
  cvt(hf, hb, (size_t)M * 2048);
  gemm(EPI_HALF, hb, w2b, ffn1_b2, x, outA, M, 512, 2048);

  // ---- rel-pos MHSA ----
  ln_kernel<512, false><<<M, 256, 0, stream>>>(outA, lnq_g, lnq_b, abf, nullptr);
  gemm(EPI_NONE, abf, wqb, nullptr, nullptr, qf, M, 512, 512);
  ln_kernel<512, false><<<M, 256, 0, stream>>>(outA, lnk_g, lnk_b, abf, nullptr);
  gemm(EPI_NONE, abf, wkb, nullptr, nullptr, kf, M, 512, 512);
  ln_kernel<512, false><<<M, 256, 0, stream>>>(outA, lnv_g, lnv_b, abf, nullptr);
  gemm(EPI_NONE, abf, wvb, nullptr, nullptr, vf, M, 512, 512);
  gemm(EPI_NONE, pebf, wpb, nullptr, nullptr, pf, 1024, 512, 512);

  {
    const size_t n = (size_t)M * 512;
    heads_kernel<<<dim3((unsigned)((n + 255) / 256)), 256, 0, stream>>>(qf, pbu, qub, pbv, qvb, n);
    heads_kernel<<<dim3((unsigned)((n + 255) / 256)), 256, 0, stream>>>(kf, nullptr, kbb, nullptr, nullptr, n);
    headsT_kernel<<<dim3((unsigned)((n + 255) / 256)), 256, 0, stream>>>(vf, vbb, n);
    const size_t np = (size_t)1024 * 512;
    heads_kernel<<<dim3((unsigned)((np + 255) / 256)), 256, 0, stream>>>(pf, nullptr, pbb, nullptr, nullptr, np);
  }
  attn_kernel<<<dim3(CT / 64, CB * CH), 128, 0, stream>>>(qub, qvb, kbb, vbb, pbb, aob);
  gemm(EPI_ADD, aob, wfb, nullptr, outA, outB, M, 512, 512);

  // ---- convolution module ----
  ln_kernel<512, false><<<M, 256, 0, stream>>>(outB, cln_g, cln_b, abf, nullptr);
  gemm(EPI_NONE, abf, pw1b, nullptr, nullptr, hf, M, 2048, 512);
  {
    const size_t n = (size_t)M * 1024;
    glu_kernel<<<dim3((unsigned)((n + 255) / 256)), 256, 0, stream>>>(hf, gluf, n);
    dwconv_kernel<<<dim3((unsigned)(n / 256)), 256, 0, stream>>>(gluf, dw_w, dcf);
  }
  ln_kernel<1024, true><<<M, 256, 0, stream>>>(dcf, bn_g, bn_b, cbb, nullptr);
  gemm(EPI_ADD, cbb, pw2b, nullptr, outB, outA, M, 512, 1024);

  // ---- FFN2 (half-step) ----
  ln_kernel<512, false><<<M, 256, 0, stream>>>(outA, ln2_g, ln2_b, abf, nullptr);
  gemm(EPI_SILU, abf, f21b, ffn2_b1, nullptr, hf, M, 2048, 512);
  cvt(hf, hb, (size_t)M * 2048);
  gemm(EPI_HALF, hb, f22b, ffn2_b2, outA, outB, M, 512, 2048);

  // ---- final LN -> d_out (f32) ----
  ln_kernel<512, false><<<M, 256, 0, stream>>>(outB, lnf_g, lnf_b, nullptr, (float*)d_out);
}
